// propagation_backward_29128468201847
// MI455X (gfx1250) — matrix operand order repaired and checked by reading, not compiled, not run on hardware
//
#include <hip/hip_runtime.h>
#include <cstdint>

// ---------------------------------------------------------------------------
// Shapes (fixed by the reference): b=2, t=8, c=64, h=w=160
// ---------------------------------------------------------------------------
constexpr int TT  = 8;
constexpr int CC  = 64;
constexpr int C2  = 128;          // 2c
constexpr int C4  = 256;          // 4c
constexpr int HH  = 160;
constexpr int WW  = 160;
constexpr int HWs = HH * WW;      // 25600
constexpr int NPIX = 2 * HWs;     // b * H * W = 51200
constexpr int KTOT = C2 * 9;      // 1152, K of the 3x3 conv GEMM
constexpr int TWc  = 32;          // pixels per conv block
constexpr int PWc  = TWc + 2;     // patch width with halo

typedef __attribute__((ext_vector_type(16))) _Float16 v16h;
typedef __attribute__((ext_vector_type(8)))  float    v8f;

union Frag16 { v16h h; uint32_t u[8]; };
struct U4 { uint32_t x, y, z, w; };

// WMMA fragment index helpers (per CDNA5 ISA 7.12.2 tables, wave32):
//  A (16-bit, 16x32): lane -> M = lane%16 ; VGPR v -> Kpair base
//      kb = (v/4)*16 + (lane/16)*8 + (v%4)*2
//  B (16-bit, 32x16): lane -> N = lane%16 ; VGPR v -> Kpair base
//      kb = (v/4)*16 + (lane/16)*8 + (v%4)*2
//  C/D (f32 16x16):   lane -> N = lane%16 ; VGPR v -> M = v + 8*(lane/16)
__device__ __forceinline__ int a_kb(int v, int half) {
    return ((v >> 2) << 4) + (half << 3) + ((v & 3) << 1);
}
__device__ __forceinline__ int b_kb(int v, int half) {
    return ((v >> 2) << 4) + (half << 3) + ((v & 3) << 1);
}

// ---------------------------------------------------------------------------
// Weight packing / conversion kernels (run once per launch)
// ---------------------------------------------------------------------------

// conv1_w (128,128,3,3) f32 -> wpk[o][k], k = (kh*3+kw)*128 + cin, f16
__global__ void k_pack_conv1(const float* __restrict__ w, _Float16* __restrict__ wpk) {
    int i = blockIdx.x * 256 + threadIdx.x;
    if (i >= C2 * KTOT) return;
    int o = i / KTOT, k = i % KTOT;
    int s = k / C2, ci = k % C2;
    int kh = s / 3, kw = s % 3;
    wpk[i] = (_Float16)w[((o * C2 + ci) * 3 + kh) * 3 + kw];
}

// plain f32 -> f16 convert (row-major weights already have K innermost)
__global__ void k_cvt16(const float* __restrict__ src, _Float16* __restrict__ dst, int n) {
    int i = blockIdx.x * 256 + threadIdx.x;
    if (i < n) dst[i] = (_Float16)src[i];
}

// copy feature[:, T-1] into d_out[:, T-1]
__global__ void k_copy_init(const float* __restrict__ feat, float* __restrict__ out) {
    size_t i = (size_t)blockIdx.x * 256 + threadIdx.x;   // over b*c*HW
    if (i >= (size_t)2 * CC * HWs) return;
    size_t rest = i % ((size_t)CC * HWs);
    int bi = (int)(i / ((size_t)CC * HWs));
    size_t off = ((size_t)bi * TT + (TT - 1)) * CC * HWs + rest;
    out[off] = feat[off];
}

// ---------------------------------------------------------------------------
// Per-step small kernels
// ---------------------------------------------------------------------------

// global average pool over H,W of feature[:, f]  -> pool[b*64]
__global__ void k_pool(const float* __restrict__ feat, int f, float* __restrict__ pool) {
    __shared__ float red[256];
    int bc = blockIdx.x;                 // 0..127
    int bi = bc >> 6, ch = bc & 63;
    const float* src = feat + (((size_t)bi * TT + f) * CC + ch) * HWs;
    float s = 0.f;
    for (int i = threadIdx.x; i < HWs; i += 256) s += src[i];
    red[threadIdx.x] = s;
    __syncthreads();
    for (int d = 128; d > 0; d >>= 1) {
        if (threadIdx.x < d) red[threadIdx.x] += red[threadIdx.x + d];
        __syncthreads();
    }
    if (threadIdx.x == 0) pool[bc] = red[0] * (1.f / (float)HWs);
}

// dynamic weight MLP: z = relu(BN(pool @ w1^T)); wd = z @ w2^T + b2
__global__ void k_dynw(const float* __restrict__ pool,
                       const float* __restrict__ w1, const float* __restrict__ gam,
                       const float* __restrict__ bet, const float* __restrict__ mu,
                       const float* __restrict__ var, const float* __restrict__ w2,
                       const float* __restrict__ b2, float* __restrict__ wd) {
    __shared__ float zs[32];
    int tid = threadIdx.x;
    if (tid < 32) {
        int bi = tid >> 4, j = tid & 15;
        float z = 0.f;
        for (int c = 0; c < CC; ++c) z += pool[bi * CC + c] * w1[j * CC + c];
        z = (z - mu[j]) * rsqrtf(var[j] + 1e-5f) * gam[j] + bet[j];
        zs[tid] = z > 0.f ? z : 0.f;
    }
    __syncthreads();
    for (int e = tid; e < 2 * CC * 9; e += 256) {
        int bi = e / (CC * 9), o = e % (CC * 9);
        float a = b2[o];
        for (int j = 0; j < 16; ++j) a += zs[bi * 16 + j] * w2[o * 16 + j];
        wd[e] = a;
    }
}

// x_feat (NCHW f32) -> fusion NHWC f16 channels [0,64)
__global__ void k_pack_x(const float* __restrict__ feat, int f, _Float16* __restrict__ fus) {
    size_t i = (size_t)blockIdx.x * 256 + threadIdx.x;   // over b*64*HW
    if (i >= (size_t)2 * CC * HWs) return;
    int yx = (int)(i % HWs);
    size_t r = i / HWs;
    int ch = (int)(r % CC);
    int bi = (int)(r / CC);
    int y = yx / WW, x = yx % WW;
    float v = feat[(((size_t)bi * TT + f) * CC + ch) * HWs + yx];
    fus[(((size_t)bi * HH + y) * WW + x) * C2 + ch] = (_Float16)v;
}

// dynamic depthwise 3x3 on feat_prop (= d_out[:, f+1], NCHW f32)
// result written as fusion NHWC f16 channels [64,128)
__global__ void k_dyndw(const float* __restrict__ out, int fprev,
                        const float* __restrict__ wd, const float* __restrict__ kcb,
                        _Float16* __restrict__ fus) {
    size_t i = (size_t)blockIdx.x * 256 + threadIdx.x;   // over b*64*HW
    if (i >= (size_t)2 * CC * HWs) return;
    int yx = (int)(i % HWs);
    size_t r = i / HWs;
    int ch = (int)(r % CC);
    int bi = (int)(r / CC);
    int y = yx / WW, x = yx % WW;
    const float* fp = out + (((size_t)bi * TT + fprev) * CC + ch) * HWs;
    const float* wv = wd + ((size_t)bi * CC + ch) * 9;
    float acc = kcb[ch];
    #pragma unroll
    for (int kh = 0; kh < 3; ++kh)
        #pragma unroll
        for (int kw = 0; kw < 3; ++kw) {
            int gy = y + kh - 1, gx = x + kw - 1;
            if ((unsigned)gy < (unsigned)HH && (unsigned)gx < (unsigned)WW)
                acc += fp[gy * WW + gx] * wv[kh * 3 + kw];
        }
    fus[(((size_t)bi * HH + y) * WW + x) * C2 + CC + ch] = (_Float16)acc;
}

// ---------------------------------------------------------------------------
// 3x3 conv 128->128 + leaky_relu(0.1) via WMMA implicit GEMM
//   grid (W/32, H, b), block 256 (8 waves); wave w owns cout tile w*16 x 32 px
//   Patch staged memory->LDS with GLOBAL_LOAD_ASYNC_TO_LDS_B128 (ASYNCcnt).
// ---------------------------------------------------------------------------
__global__ __launch_bounds__(256) void k_conv3x3(
    const _Float16* __restrict__ fus,   // NHWC (b,H,W,128) f16
    const _Float16* __restrict__ wpk,   // [128][1152] f16, K = (kh*3+kw)*128+ci
    const float*    __restrict__ bias,  // [128]
    float*          __restrict__ f12f,  // NCHW (b,128,H,W) f32
    _Float16*       __restrict__ f12h)  // NHWC (b,H,W,128) f16
{
    __shared__ _Float16 patch[3 * PWc * C2];   // 26112 B
    const int x0 = blockIdx.x * TWc;
    const int y  = blockIdx.y;
    const int bi = blockIdx.z;
    const int tid = threadIdx.x;

    // --- stage halo patch: rows y-1..y+1, cols x0-1..x0+32, 128 channels ---
    // 16-byte granules: per (row,col) the 128 f16 channels are 16 granules.
    const int NG = 3 * PWc * (C2 / 8);          // 1632 granules of 16 B
    // 1) zero-fill (uniformly handles SAME padding halo)
    for (int e = tid; e < NG; e += 256)
        ((U4*)patch)[e] = U4{0u, 0u, 0u, 0u};
    __syncthreads();
    // 2) async copy in-bounds granules directly memory -> LDS (no VGPR bounce)
    for (int e = tid; e < NG; e += 256) {
        int chunk = e & 15;                     // 16B granule within 128 ch
        int colr  = e >> 4;
        int col   = colr % PWc;
        int rr    = colr / PWc;
        int gy = y + rr - 1;
        int gx = x0 + col - 1;
        if ((unsigned)gy < (unsigned)HH && (unsigned)gx < (unsigned)WW) {
            const _Float16* gp =
                fus + (((size_t)bi * HH + gy) * WW + gx) * C2 + chunk * 8;
            uint32_t lds_off =
                (uint32_t)(size_t)(patch + ((size_t)(rr * PWc + col)) * C2 + chunk * 8);
            asm volatile("global_load_async_to_lds_b128 %0, %1, off"
                         :: "v"(lds_off), "v"(gp) : "memory");
        }
    }
    asm volatile("s_wait_asynccnt 0x0" ::: "memory");
    __syncthreads();

    const int wave = tid >> 5, lane = tid & 31;
    const int m0 = wave * 16;
    const int half = lane >> 4, n = lane & 15;
    const _Float16* arow = wpk + (size_t)(m0 + n) * KTOT;
    __builtin_prefetch(arow, 0, 0);             // global_prefetch toward WGP

    v8f acc0 = {}; v8f acc1 = {};
    for (int k0 = 0; k0 < KTOT; k0 += 32) {
        Frag16 a, b0, b1;
        const int s  = k0 >> 7;          // spatial tap 0..8
        const int c0 = k0 & (C2 - 1);    // channel base within tap
        const int kh = s / 3, kw = s - kh * 3;
        const _Float16* bb0 = patch + ((kh * PWc + (n + kw)) * C2 + c0);
        const _Float16* bb1 = patch + ((kh * PWc + (16 + n + kw)) * C2 + c0);
        #pragma unroll
        for (int v = 0; v < 8; ++v) {
            a.u[v]  = *(const uint32_t*)(arow + k0 + a_kb(v, half));
            b0.u[v] = *(const uint32_t*)(bb0 + b_kb(v, half));
            b1.u[v] = *(const uint32_t*)(bb1 + b_kb(v, half));
        }
        acc0 = __builtin_amdgcn_wmma_f32_16x16x32_f16(false, a.h, false, b0.h,
                                                      (short)0, acc0, false, false);
        acc1 = __builtin_amdgcn_wmma_f32_16x16x32_f16(false, a.h, false, b1.h,
                                                      (short)0, acc1, false, false);
    }

    #pragma unroll
    for (int v = 0; v < 8; ++v) {
        const int m = m0 + v + 8 * half;
        const float bv = bias[m];
        float v0 = acc0[v] + bv; v0 = v0 > 0.f ? v0 : 0.1f * v0;
        float v1 = acc1[v] + bv; v1 = v1 > 0.f ? v1 : 0.1f * v1;
        const int xA = x0 + n, xB = x0 + 16 + n;
        f12f[((size_t)bi * C2 + m) * HWs + y * WW + xA] = v0;
        f12f[((size_t)bi * C2 + m) * HWs + y * WW + xB] = v1;
        f12h[(((size_t)bi * HH + y) * WW + xA) * C2 + m] = (_Float16)v0;
        f12h[(((size_t)bi * HH + y) * WW + xB) * C2 + m] = (_Float16)v1;
    }
}

// ---------------------------------------------------------------------------
// Pointwise (1x1) conv as WMMA GEMM, fragments straight from global NHWC f16.
//   act==0 : out16[p*M + m] = acc + bias            (pi: 64 -> 256)
//   act==1 : outf NCHW      = sigmoid(acc + bias)   (po: 256 -> 64)
//   launch: blockDim = min(M/16,8)*32 ; grid = (NPIX/32, M/(8*16) or 1)
// ---------------------------------------------------------------------------
__global__ void k_pw(const _Float16* __restrict__ X, int ldx, int xoff,
                     const _Float16* __restrict__ Wp, const float* __restrict__ bias,
                     int K, int M,
                     _Float16* __restrict__ out16, float* __restrict__ outf, int act)
{
    const int tid = threadIdx.x;
    const int wave = tid >> 5, lane = tid & 31;
    const int m0 = (blockIdx.y * (blockDim.x >> 5) + wave) * 16;
    const int p0 = blockIdx.x * 32;
    const int half = lane >> 4, n = lane & 15;
    const _Float16* arow = Wp + (size_t)(m0 + n) * K;
    const _Float16* xr0 = X + (size_t)(p0 + n) * ldx + xoff;
    const _Float16* xr1 = X + (size_t)(p0 + 16 + n) * ldx + xoff;
    __builtin_prefetch(arow, 0, 0);

    v8f acc0 = {}; v8f acc1 = {};
    for (int k0 = 0; k0 < K; k0 += 32) {
        Frag16 a, b0, b1;
        #pragma unroll
        for (int v = 0; v < 8; ++v) {
            a.u[v]  = *(const uint32_t*)(arow + k0 + a_kb(v, half));
            b0.u[v] = *(const uint32_t*)(xr0 + k0 + b_kb(v, half));
            b1.u[v] = *(const uint32_t*)(xr1 + k0 + b_kb(v, half));
        }
        acc0 = __builtin_amdgcn_wmma_f32_16x16x32_f16(false, a.h, false, b0.h,
                                                      (short)0, acc0, false, false);
        acc1 = __builtin_amdgcn_wmma_f32_16x16x32_f16(false, a.h, false, b1.h,
                                                      (short)0, acc1, false, false);
    }

    #pragma unroll
    for (int v = 0; v < 8; ++v) {
        const int m = m0 + v + 8 * half;
        const float bv = bias[m];
        float v0 = acc0[v] + bv;
        float v1 = acc1[v] + bv;
        const int pA = p0 + n, pB = p0 + 16 + n;
        if (act == 1) {
            v0 = 1.f / (1.f + __expf(-v0));
            v1 = 1.f / (1.f + __expf(-v1));
            int bA = pA / HWs, yxA = pA % HWs;
            int bB = pB / HWs, yxB = pB % HWs;
            outf[((size_t)bA * M + m) * HWs + yxA] = v0;
            outf[((size_t)bB * M + m) * HWs + yxB] = v1;
        } else {
            out16[(size_t)pA * M + m] = (_Float16)v0;
            out16[(size_t)pB * M + m] = (_Float16)v1;
        }
    }
}

// depthwise 3x3 on NHWC f16, Cn channels (one thread per channel, one block per pixel)
__global__ void k_dw(const _Float16* __restrict__ X, const float* __restrict__ Wd,
                     const float* __restrict__ bd, _Float16* __restrict__ Y, int Cn)
{
    const int p = blockIdx.x;             // 0 .. b*H*W-1
    const int ch = threadIdx.x;           // Cn threads
    const int bi = p / HWs, yx = p % HWs;
    const int y = yx / WW, x = yx % WW;
    float acc = bd[ch];
    #pragma unroll
    for (int kh = 0; kh < 3; ++kh)
        #pragma unroll
        for (int kw = 0; kw < 3; ++kw) {
            int gy = y + kh - 1, gx = x + kw - 1;
            if ((unsigned)gy < (unsigned)HH && (unsigned)gx < (unsigned)WW)
                acc += (float)X[(((size_t)bi * HH + gy) * WW + gx) * Cn + ch]
                       * Wd[ch * 9 + kh * 3 + kw];
        }
    Y[(size_t)p * Cn + ch] = (_Float16)acc;
}

// new_fp = f1*g1 + f2*g2  -> d_out[:, frame]
__global__ void k_combine(const float* __restrict__ f12f, const float* __restrict__ g1,
                          const float* __restrict__ g2, float* __restrict__ out, int frame)
{
    size_t i = (size_t)blockIdx.x * 256 + threadIdx.x;   // over b*64*HW
    if (i >= (size_t)2 * CC * HWs) return;
    int yx = (int)(i % HWs);
    size_t r = i / HWs;
    int ch = (int)(r % CC);
    int bi = (int)(r / CC);
    float f1 = f12f[((size_t)bi * C2 + ch) * HWs + yx];
    float f2 = f12f[((size_t)bi * C2 + CC + ch) * HWs + yx];
    float v = f1 * g1[((size_t)bi * CC + ch) * HWs + yx]
            + f2 * g2[((size_t)bi * CC + ch) * HWs + yx];
    out[(((size_t)bi * TT + frame) * CC + ch) * HWs + yx] = v;
}

// ---------------------------------------------------------------------------
// Host orchestration
// ---------------------------------------------------------------------------
extern "C" void kernel_launch(void* const* d_in, const int* in_sizes, int n_in,
                              void* d_out, int out_size, void* d_ws, size_t ws_size,
                              hipStream_t stream) {
    const float* feature = (const float*)d_in[0];
    const float* kc_w1   = (const float*)d_in[1];
    const float* kc_g    = (const float*)d_in[2];
    const float* kc_b    = (const float*)d_in[3];
    const float* kc_mu   = (const float*)d_in[4];
    const float* kc_var  = (const float*)d_in[5];
    const float* kc_w2   = (const float*)d_in[6];
    const float* kc_b2   = (const float*)d_in[7];
    const float* kc_bias = (const float*)d_in[8];
    const float* conv1_w = (const float*)d_in[9];
    const float* conv1_b = (const float*)d_in[10];
    const float* pi1_w   = (const float*)d_in[11];
    const float* pi1_b   = (const float*)d_in[12];
    const float* dw1_w   = (const float*)d_in[13];
    const float* dw1_b   = (const float*)d_in[14];
    const float* po1_w   = (const float*)d_in[15];
    const float* po1_b   = (const float*)d_in[16];
    const float* pi2_w   = (const float*)d_in[17];
    const float* pi2_b   = (const float*)d_in[18];
    const float* dw2_w   = (const float*)d_in[19];
    const float* dw2_b   = (const float*)d_in[20];
    const float* po2_w   = (const float*)d_in[21];
    const float* po2_b   = (const float*)d_in[22];
    float* out = (float*)d_out;

    // workspace carve-up (256-B aligned)
    char* ws = (char*)d_ws;
    size_t off = 0;
    auto alloc = [&](size_t bytes) {
        void* p = ws + off;
        off += (bytes + 255) & ~(size_t)255;
        return p;
    };
    _Float16* wpk1   = (_Float16*)alloc((size_t)C2 * KTOT * 2);         // 288 KB
    _Float16* pi1pk  = (_Float16*)alloc((size_t)C4 * CC * 2);
    _Float16* po1pk  = (_Float16*)alloc((size_t)CC * C4 * 2);
    _Float16* pi2pk  = (_Float16*)alloc((size_t)C4 * CC * 2);
    _Float16* po2pk  = (_Float16*)alloc((size_t)CC * C4 * 2);
    _Float16* fus16  = (_Float16*)alloc((size_t)NPIX * C2 * 2);         // 12.5 MB
    float*    f12f   = (float*)   alloc((size_t)NPIX * C2 * 4);         // 25 MB
    _Float16* f12h   = (_Float16*)alloc((size_t)NPIX * C2 * 2);         // 12.5 MB
    _Float16* t1     = (_Float16*)alloc((size_t)NPIX * C4 * 2);         // 25 MB
    _Float16* t2     = (_Float16*)alloc((size_t)NPIX * C4 * 2);         // 25 MB
    float*    g1     = (float*)   alloc((size_t)NPIX * CC * 4);         // 12.5 MB
    float*    g2     = (float*)   alloc((size_t)NPIX * CC * 4);         // 12.5 MB
    float*    pool   = (float*)   alloc(2 * CC * 4);
    float*    wd     = (float*)   alloc(2 * CC * 9 * 4);
    (void)ws_size; (void)n_in; (void)in_sizes; (void)out_size;

    // one-time per launch: pack weights to f16, seed frame T-1
    k_pack_conv1<<<(C2 * KTOT + 255) / 256, 256, 0, stream>>>(conv1_w, wpk1);
    k_cvt16<<<(C4 * CC + 255) / 256, 256, 0, stream>>>(pi1_w, pi1pk, C4 * CC);
    k_cvt16<<<(C4 * CC + 255) / 256, 256, 0, stream>>>(po1_w, po1pk, C4 * CC);
    k_cvt16<<<(C4 * CC + 255) / 256, 256, 0, stream>>>(pi2_w, pi2pk, C4 * CC);
    k_cvt16<<<(C4 * CC + 255) / 256, 256, 0, stream>>>(po2_w, po2pk, C4 * CC);

    const int gElem = (2 * CC * HWs + 255) / 256;   // 12800 blocks
    k_copy_init<<<gElem, 256, 0, stream>>>(feature, out);

    // reverse temporal scan: produce frames T-2 .. 0; feat_prop = d_out[:, f+1]
    for (int s = 0; s < TT - 1; ++s) {
        const int f = TT - 2 - s;

        k_pool<<<2 * CC, 256, 0, stream>>>(feature, f, pool);
        k_dynw<<<1, 256, 0, stream>>>(pool, kc_w1, kc_g, kc_b, kc_mu, kc_var,
                                      kc_w2, kc_b2, wd);
        k_pack_x<<<gElem, 256, 0, stream>>>(feature, f, fus16);
        k_dyndw<<<gElem, 256, 0, stream>>>(out, f + 1, wd, kc_bias, fus16);

        k_conv3x3<<<dim3(WW / TWc, HH, 2), 256, 0, stream>>>(fus16, wpk1, conv1_b,
                                                             f12f, f12h);
        // branch 1: g1 = sigmoid(po1(dw1(pi1(f1))))
        k_pw<<<dim3(NPIX / 32, 2), 256, 0, stream>>>(f12h, C2, 0,  pi1pk, pi1_b,
                                                     CC, C4, t1, nullptr, 0);
        k_dw<<<NPIX, C4, 0, stream>>>(t1, dw1_w, dw1_b, t2, C4);
        k_pw<<<dim3(NPIX / 32, 1), 128, 0, stream>>>(t2, C4, 0,  po1pk, po1_b,
                                                     C4, CC, nullptr, g1, 1);
        // branch 2: g2 = sigmoid(po2(dw2(pi2(f2))))
        k_pw<<<dim3(NPIX / 32, 2), 256, 0, stream>>>(f12h, C2, CC, pi2pk, pi2_b,
                                                     CC, C4, t1, nullptr, 0);
        k_dw<<<NPIX, C4, 0, stream>>>(t1, dw2_w, dw2_b, t2, C4);
        k_pw<<<dim3(NPIX / 32, 1), 128, 0, stream>>>(t2, C4, 0,  po2pk, po2_b,
                                                     C4, CC, nullptr, g2, 1);

        k_combine<<<gElem, 256, 0, stream>>>(f12f, g1, g2, out, f);
    }
}
